// MultiHeadSelfAttention_2413771620711
// MI455X (gfx1250) — matrix operand order repaired and checked by reading, not compiled, not run on hardware
//
#include <hip/hip_runtime.h>

typedef __bf16 bf16;
typedef __attribute__((ext_vector_type(16))) __bf16 v16bf;
typedef __attribute__((ext_vector_type(8)))  __bf16 v8bf;
typedef __attribute__((ext_vector_type(4)))  __bf16 v4bf;
typedef __attribute__((ext_vector_type(8)))  float  v8f;

#define D_MODEL 1024
#define SEQ     2048
#define NHEAD   16
#define DHEAD   64
#define BATCH   2
#define LDSPAD  40   // row stride (elements): 80B -> L*20 mod 64 banks, conflict-free

// ---------------------------------------------------------------------------
// WMMA fragment helpers (wave32 layouts per CDNA5 ISA 7.12.2)
// ---------------------------------------------------------------------------

// Build a 16-element bf16 fragment from two 16B chunks.
__device__ inline v16bf make_frag(const bf16* lo_p, const bf16* hi_p) {
    v8bf lo = *(const v8bf*)lo_p;
    v8bf hi = *(const v8bf*)hi_p;
    v16bf a;
#pragma unroll
    for (int i = 0; i < 8; ++i) { a[i] = lo[i]; a[i + 8] = hi[i]; }
    return a;
}

// A-matrix 16x32 bf16 fragment. rowPtr = &A[m * ld + k] for lane's row m.
// element j -> k = half*16 + j: chunks at +half*16, +half*16+8.
__device__ inline v16bf load_a_frag(const bf16* rowPtr) {
    const int half = (threadIdx.x >> 4) & 1;
    return make_frag(rowPtr + half * 16, rowPtr + half * 16 + 8);
}

// B-matrix 32x16 fragment: lane holds col n; lanes 0-15 K=0..15, 16-31 K=16..31.
// colPtr = &B[n*ld + k + half*16]: 16 contiguous bf16.
__device__ inline v16bf load_b_frag(const bf16* colPtr) {
    return make_frag(colPtr, colPtr + 8);
}

__device__ inline v8f wmma_bf16(v16bf a, v16bf b, v8f c) {
    return __builtin_amdgcn_wmma_f32_16x16x32_bf16(false, a, false, b,
                                                   (short)0, c, false, false);
}

__device__ inline float rowmax16(float v) {
#pragma unroll
    for (int m = 1; m < 16; m <<= 1) v = fmaxf(v, __shfl_xor(v, m, 32));
    return v;
}
__device__ inline float rowsum16(float v) {
#pragma unroll
    for (int m = 1; m < 16; m <<= 1) v += __shfl_xor(v, m, 32);
    return v;
}

// ---------------------------------------------------------------------------
// CDNA5 async global->LDS copy (16B per lane), tracked with ASYNCcnt.
// Low 32 bits of a generic pointer to __shared__ are the LDS byte offset.
// ---------------------------------------------------------------------------
__device__ inline void async_b128(const bf16* gbase, int gbyte, void* lds) {
    unsigned lds_off = (unsigned)(size_t)lds;
    asm volatile("global_load_async_to_lds_b128 %0, %1, %2"
                 :: "v"(lds_off), "v"(gbyte), "s"(gbase)
                 : "memory");
}

// ---------------------------------------------------------------------------
// fp32 -> bf16 conversion
// ---------------------------------------------------------------------------
__global__ void cvt_f32_bf16_kernel(const float* __restrict__ src,
                                    bf16* __restrict__ dst, int n) {
    int i = (blockIdx.x * blockDim.x + threadIdx.x) * 4;
    if (i < n) {
        float4 f = *(const float4*)(src + i);
        v4bf o;
        o[0] = (bf16)f.x; o[1] = (bf16)f.y; o[2] = (bf16)f.z; o[3] = (bf16)f.w;
        *(v4bf*)(dst + i) = o;
    }
}

// ---------------------------------------------------------------------------
// Staged block GEMM: 256 threads (8 waves) compute a 128x64 tile of A*B^T.
// A: [M x K] row-major bf16, B: [N x K] row-major bf16 (weights are (d_out,d_in)).
// Double-buffered async LDS staging with COMPILE-TIME buffer parity (K stages
// processed in pairs) so accumulators stay register-tied (D==C for v_wmma).
// Per stage each thread issues 3x global_load_async_to_lds_b128; the next
// stage is issued before waiting on the current one (s_wait_asynccnt 3).
// ---------------------------------------------------------------------------
__device__ __forceinline__ void block_gemm_128x64(const bf16* __restrict__ A,
                                                  const bf16* __restrict__ B,
                                                  const int K, const int mTile,
                                                  const int nTile, v8f acc[4]) {
    __shared__ bf16 As[2][128][LDSPAD];
    __shared__ bf16 Bs[2][64][LDSPAD];

    const int t    = threadIdx.x;
    const int lane = t & 31;
    const int wave = t >> 5;
    const int half = lane >> 4;
    const int l16  = lane & 15;
    const int row  = wave * 16 + l16;

    // per-thread copy coordinates: A = 512 chunks (2/thread), B = 256 (1/thread)
    const int rC = t >> 2;            // 0..63
    const int oC = (t & 3) * 8;       // element offset within 32-wide row

    const int gA0 = (mTile + rC) * K + oC;        // element offsets
    const int gA1 = (mTile + rC + 64) * K + oC;
    const int gB0 = (nTile + rC) * K + oC;

    auto issue = [&](int kb, int buf) {
        const int k = kb * 32;
        async_b128(A, (gA0 + k) * 2, &As[buf][rC][oC]);
        async_b128(A, (gA1 + k) * 2, &As[buf][rC + 64][oC]);
        async_b128(B, (gB0 + k) * 2, &Bs[buf][rC][oC]);
    };
    auto compute = [&](int buf) {
        v16bf a = make_frag(&As[buf][row][half * 16], &As[buf][row][half * 16 + 8]);
#pragma unroll
        for (int tt = 0; tt < 4; ++tt) {
            const bf16* bp = &Bs[buf][tt * 16 + l16][half * 16];
            acc[tt] = wmma_bf16(a, make_frag(bp, bp + 8), acc[tt]);
        }
    };

    const int nK = K / 32;            // even (K = 1024 -> 32 stages)
    issue(0, 0);
    for (int kb = 0; kb < nK; kb += 2) {
        // ---- even stage: compute buf 0, prefetch stage kb+1 into buf 1 ----
        issue(kb + 1, 1);
        asm volatile("s_wait_asynccnt 3" ::: "memory");   // stage kb in LDS
        __syncthreads();
        compute(0);
        __syncthreads();                                  // readers done w/ buf0
        // ---- odd stage: compute buf 1, prefetch stage kb+2 into buf 0 ----
        if (kb + 2 < nK) {
            issue(kb + 2, 0);
            asm volatile("s_wait_asynccnt 3" ::: "memory");
        } else {
            asm volatile("s_wait_asynccnt 0" ::: "memory");
        }
        __syncthreads();
        compute(1);
        __syncthreads();                                  // readers done w/ buf1
    }
}

// ---------------------------------------------------------------------------
// QKV projection: grid.z selects Q(0)/K(1)/V(2).
// Q,K -> [b][h][s][dh]; V -> transposed [b][h][dh][s].
// ---------------------------------------------------------------------------
__global__ __launch_bounds__(256)
void gemm_qkv_kernel(const bf16* __restrict__ X,
                     const bf16* __restrict__ Wq, const bf16* __restrict__ Wk,
                     const bf16* __restrict__ Wv,
                     bf16* __restrict__ Qo, bf16* __restrict__ Ko,
                     bf16* __restrict__ Vto) {
    const int which = blockIdx.z;
    const bf16* W = (which == 0) ? Wq : (which == 1) ? Wk : Wv;
    const int lane = threadIdx.x & 31;
    const int wave = threadIdx.x >> 5;
    const int half = lane >> 4;
    const int l16  = lane & 15;
    const int mTile = blockIdx.x * 128;
    const int nBase = blockIdx.y * 64;

    v8f acc[4] = {};
    block_gemm_128x64(X, W, D_MODEL, mTile, nBase, acc);

#pragma unroll
    for (int tt = 0; tt < 4; ++tt) {
#pragma unroll
        for (int r = 0; r < 8; ++r) {
            int m = mTile + wave * 16 + r + half * 8;
            int n = nBase + tt * 16 + l16;
            int b = m >> 11, s = m & (SEQ - 1);
            int h = n >> 6,  d = n & (DHEAD - 1);
            bf16 v = (bf16)acc[tt][r];
            if (which == 2)
                Vto[((size_t)(b * NHEAD + h) * DHEAD + d) * SEQ + s] = v;
            else {
                bf16* dst = which ? Ko : Qo;
                dst[((size_t)(b * NHEAD + h) * SEQ + s) * DHEAD + d] = v;
            }
        }
    }
}

// ---------------------------------------------------------------------------
// Causal flash attention. Block = 4 waves; wave owns 16 query rows and runs
// an independent online-softmax loop over KV chunks of 32.
// ---------------------------------------------------------------------------
__global__ __launch_bounds__(128)
void attn_kernel(const bf16* __restrict__ Q, const bf16* __restrict__ K,
                 const bf16* __restrict__ Vt, bf16* __restrict__ attn) {
    const int qTile = blockIdx.x;
    const int h = blockIdx.y, b = blockIdx.z;
    const int wave = threadIdx.x >> 5;
    const int lane = threadIdx.x & 31;
    const int half = lane >> 4;
    const int l16  = lane & 15;
    const int qRow0 = qTile * 64 + wave * 16;

    const size_t head = (size_t)(b * NHEAD + h);
    const bf16* Qh = Q  + head * SEQ * DHEAD;   // [S][64]
    const bf16* Kh = K  + head * SEQ * DHEAD;   // [S][64]
    const bf16* Vh = Vt + head * DHEAD * SEQ;   // [64][S]

    const bf16* qp = Qh + (size_t)(qRow0 + l16) * DHEAD;
    const v16bf Qf0 = load_a_frag(qp);
    const v16bf Qf1 = load_a_frag(qp + 32);

    float mstat[8], lstat[8];
#pragma unroll
    for (int r = 0; r < 8; ++r) { mstat[r] = -3.0e38f; lstat[r] = 0.0f; }
    v8f Of[4] = {};

    __shared__ bf16 Plds[4][16][32];
    bf16 (*P)[32] = Plds[wave];

    const float scale = 0.125f;                 // 1/sqrt(64)
    const int nChunks = ((qRow0 + 15) >> 5) + 1;

    for (int c = 0; c < nChunks; ++c) {
        const int kv0 = c * 32;
        v8f S0 = {}, S1 = {};
        {
            const bf16* kp = Kh + (size_t)(kv0 + l16) * DHEAD + half * 16;
            S0 = wmma_bf16(Qf0, load_b_frag(kp), S0);
            S0 = wmma_bf16(Qf1, load_b_frag(kp + 32), S0);
        }
        {
            const bf16* kp = Kh + (size_t)(kv0 + 16 + l16) * DHEAD + half * 16;
            S1 = wmma_bf16(Qf0, load_b_frag(kp), S1);
            S1 = wmma_bf16(Qf1, load_b_frag(kp + 32), S1);
        }
        float alpha[8];
#pragma unroll
        for (int r = 0; r < 8; ++r) {
            const int row = qRow0 + r + half * 8;
            float v0 = S0[r] * scale; if (kv0 + l16 > row)      v0 = -1.0e9f;
            float v1 = S1[r] * scale; if (kv0 + 16 + l16 > row) v1 = -1.0e9f;
            float mx = rowmax16(fmaxf(v0, v1));
            float mnew = fmaxf(mstat[r], mx);
            float p0 = __expf(v0 - mnew);
            float p1 = __expf(v1 - mnew);
            float ps = rowsum16(p0 + p1);
            alpha[r] = __expf(mstat[r] - mnew);
            lstat[r] = lstat[r] * alpha[r] + ps;
            mstat[r] = mnew;
            P[r + half * 8][l16]      = (bf16)p0;
            P[r + half * 8][16 + l16] = (bf16)p1;
        }
#pragma unroll
        for (int tt = 0; tt < 4; ++tt)
#pragma unroll
            for (int r = 0; r < 8; ++r) Of[tt][r] = Of[tt][r] * alpha[r];

        // wave-private LDS re-layout: C-frag (col-per-lane) -> A-frag (row-per-lane)
        asm volatile("s_wait_dscnt 0" ::: "memory");
        v16bf Pf = load_a_frag(&P[l16][0]);

#pragma unroll
        for (int tt = 0; tt < 4; ++tt) {
            const bf16* vp = Vh + (size_t)(tt * 16 + l16) * SEQ + kv0 + half * 16;
            Of[tt] = wmma_bf16(Pf, load_b_frag(vp), Of[tt]);
        }
    }

#pragma unroll
    for (int tt = 0; tt < 4; ++tt)
#pragma unroll
        for (int r = 0; r < 8; ++r) {
            int s = qRow0 + r + half * 8;
            int d = tt * 16 + l16;
            float o = Of[tt][r] / lstat[r];
            attn[(size_t)(b * SEQ + s) * D_MODEL + h * DHEAD + d] = (bf16)o;
        }
}

// ---------------------------------------------------------------------------
// Output projection: fp32 result to d_out.
// ---------------------------------------------------------------------------
__global__ __launch_bounds__(256)
void gemm_out_kernel(const bf16* __restrict__ A, const bf16* __restrict__ Wo,
                     float* __restrict__ out) {
    const int lane = threadIdx.x & 31;
    const int wave = threadIdx.x >> 5;
    const int half = lane >> 4;
    const int l16  = lane & 15;
    const int mTile = blockIdx.x * 128;
    const int nBase = blockIdx.y * 64;

    v8f acc[4] = {};
    block_gemm_128x64(A, Wo, D_MODEL, mTile, nBase, acc);

#pragma unroll
    for (int tt = 0; tt < 4; ++tt)
#pragma unroll
        for (int r = 0; r < 8; ++r) {
            int m = mTile + wave * 16 + r + half * 8;
            int n = nBase + tt * 16 + l16;
            out[(size_t)m * D_MODEL + n] = acc[tt][r];
        }
}

// ---------------------------------------------------------------------------
extern "C" void kernel_launch(void* const* d_in, const int* in_sizes, int n_in,
                              void* d_out, int out_size, void* d_ws, size_t ws_size,
                              hipStream_t stream) {
    const float* x  = (const float*)d_in[0];
    const float* wq = (const float*)d_in[1];
    const float* wk = (const float*)d_in[2];
    const float* wv = (const float*)d_in[3];
    const float* wo = (const float*)d_in[4];

    char* ws = (char*)d_ws;
    const size_t MiB = 1u << 20;
    bf16* xb  = (bf16*)(ws);              //  8 MiB: x bf16 [4096][1024]
    bf16* wqb = (bf16*)(ws +  8 * MiB);   //  2 MiB each
    bf16* wkb = (bf16*)(ws + 10 * MiB);
    bf16* wvb = (bf16*)(ws + 12 * MiB);
    bf16* wob = (bf16*)(ws + 14 * MiB);
    bf16* qb  = (bf16*)(ws + 16 * MiB);   //  8 MiB: [b][h][s][dh]
    bf16* kb  = (bf16*)(ws + 24 * MiB);   //  8 MiB: [b][h][s][dh]
    bf16* vtb = (bf16*)(ws + 32 * MiB);   //  8 MiB: [b][h][dh][s]
    bf16* ab  = (bf16*)(ws + 40 * MiB);   //  8 MiB: attn out [4096][1024]

    const int nX = BATCH * SEQ * D_MODEL;
    const int nW = D_MODEL * D_MODEL;

    cvt_f32_bf16_kernel<<<nX / 4 / 256, 256, 0, stream>>>(x,  xb,  nX);
    cvt_f32_bf16_kernel<<<nW / 4 / 256, 256, 0, stream>>>(wq, wqb, nW);
    cvt_f32_bf16_kernel<<<nW / 4 / 256, 256, 0, stream>>>(wk, wkb, nW);
    cvt_f32_bf16_kernel<<<nW / 4 / 256, 256, 0, stream>>>(wv, wvb, nW);
    cvt_f32_bf16_kernel<<<nW / 4 / 256, 256, 0, stream>>>(wo, wob, nW);

    dim3 gQKV(BATCH * SEQ / 128, D_MODEL / 64, 3);
    gemm_qkv_kernel<<<gQKV, 256, 0, stream>>>(xb, wqb, wkb, wvb, qb, kb, vtb);

    dim3 gAttn(SEQ / 64, NHEAD, BATCH);
    attn_kernel<<<gAttn, 128, 0, stream>>>(qb, kb, vtb, ab);

    dim3 gOut(BATCH * SEQ / 128, D_MODEL / 64);
    gemm_out_kernel<<<gOut, 256, 0, stream>>>(ab, wob, (float*)d_out);
}
